// JITODESolver_44470091383428
// MI455X (gfx1250) — matrix operand order repaired and checked by reading, not compiled, not run on hardware
//
#include <hip/hip_runtime.h>
#include <hip/hip_bf16.h>

typedef __attribute__((ext_vector_type(16))) _Float16 v16h;
typedef __attribute__((ext_vector_type(8)))  float    v8f;

#define B_    2048
#define IN_   128
#define H_    256
#define H2_   512
#define OUT_  128
#define NSTEP 64
#define MB    64      // batch rows per workgroup (4 row-tiles)

// ws byte offsets (all 32B aligned)
#define WX_OFF    0        // 4*32 frags  * 1024B = 131072
#define WH_OFF    131072   // 8*32 frags  * 1024B = 262144
#define WOUT_OFF  393216   // 8*8  frags  * 1024B = 65536
#define XF_OFF    458752   // 128*4 frags * 1024B = 524288

// ---- fragment K-index maps (ISA 7.12.2) ----
__device__ __forceinline__ int ka_of(int lane, int e) {
  int hi = lane >> 4;
  return (e < 8) ? (hi * 8 + e) : (16 + hi * 8 + (e - 8));
}
__device__ __forceinline__ int kb_of(int lane, int e) {
  int hi = lane >> 4;
  return (e < 8) ? (hi * 8 + e) : (16 + hi * 8 + (e - 8));
}

__device__ __forceinline__ v8f wmma_f16(v16h a, v16h b, v8f c) {
  return __builtin_amdgcn_wmma_f32_16x16x32_f16(false, a, false, b, (short)0, c,
                                                false, false);
}

// ---------- pack kernels ----------
// W2 = [w_gate ; w] rows, columns INTERLEAVED: n even -> gate col n/2, n odd -> f col n/2.
__global__ void pack_w2(const float* __restrict__ wg, const float* __restrict__ w,
                        _Float16* __restrict__ dst, int rowoff) {
  int t = blockIdx.x * 256 + threadIdx.x;
  int e = t & 15, lane = (t >> 4) & 31, f = t >> 9;
  int kt = f >> 5, nt = f & 31;
  int k = rowoff + kt * 32 + kb_of(lane, e);
  int n = nt * 16 + (lane & 15);
  const float* s = (n & 1) ? w : wg;
  dst[t] = (_Float16)s[k * H_ + (n >> 1)];
}

__global__ void pack_wout(const float* __restrict__ wo, _Float16* __restrict__ dst) {
  int t = blockIdx.x * 256 + threadIdx.x;
  int e = t & 15, lane = (t >> 4) & 31, f = t >> 9;
  int kt = f >> 3, nt = f & 7;
  int k = kt * 32 + kb_of(lane, e);
  int n = nt * 16 + (lane & 15);
  dst[t] = (_Float16)wo[k * OUT_ + n];
}

// x (2048x128) as A-frags: frag = rt*4 + kt
__global__ void pack_x(const float* __restrict__ x, _Float16* __restrict__ dst) {
  int t = blockIdx.x * 256 + threadIdx.x;
  int e = t & 15, lane = (t >> 4) & 31, f = t >> 9;
  int rt = f >> 2, kt = f & 3;
  int m = rt * 16 + (lane & 15);
  int k = kt * 32 + ka_of(lane, e);
  dst[t] = (_Float16)x[m * IN_ + k];
}

// ---------- fused scan kernel ----------
// 512 threads = 16 waves: wave = (rp in 0..1) x (wc in 0..7)
// Each wave: TWO 16-row tiles (rt = rp*2+u) x 64 interleaved cols; every B-frag
// load feeds two WMMAs (row-tile reuse) to halve weight-load traffic per FLOP.
__global__ void __launch_bounds__(512)
lnn_main(const _Float16* __restrict__ XF, const _Float16* __restrict__ WX,
         const _Float16* __restrict__ WH, const _Float16* __restrict__ WOUT,
         const float* __restrict__ t_eval, const float* __restrict__ b_gate,
         const float* __restrict__ b_f, const float* __restrict__ tau,
         const float* __restrict__ b_out, float* __restrict__ out) {
  __shared__ alignas(32) _Float16 lds_h[4 * 8 * 512];  // 4 row-tiles x 8 k-tiles, 32KB

  const int tid  = threadIdx.x;
  const int lane = tid & 31;
  const int wave = tid >> 5;
  const int rp   = wave >> 3;   // row-tile pair (tiles rp*2, rp*2+1)
  const int wc   = wave & 7;    // col block (64 interleaved cols)
  const int wg   = blockIdx.x;
  const int lo   = lane & 15;
  const int hi4  = lane >> 4;
  const int par  = lane & 1;    // 0: holds gate cols, 1: holds f cols

  // zero initial h fragments (16384 halves / 512 threads = 32 each)
#pragma unroll
  for (int i = 0; i < 32; ++i) lds_h[tid * 32 + i] = (_Float16)0.f;
  __syncthreads();

  const v16h* XFv = (const v16h*)XF;
  const v16h* WXv = (const v16h*)WX;
  const v16h* WOv = (const v16h*)WOUT;

  // ---- phase 1: XZ = x @ Wx (+ biases folded in) ----
  v8f XZ[2][4];
#pragma unroll
  for (int u = 0; u < 2; ++u)
#pragma unroll
    for (int n = 0; n < 4; ++n) XZ[u][n] = {};
#pragma unroll
  for (int kt = 0; kt < 4; ++kt) {
    v16h a0 = XFv[((wg * 4 + rp * 2 + 0) * 4 + kt) * 32 + lane];
    v16h a1 = XFv[((wg * 4 + rp * 2 + 1) * 4 + kt) * 32 + lane];
#pragma unroll
    for (int n = 0; n < 4; ++n) {
      v16h b = WXv[(kt * 32 + wc * 4 + n) * 32 + lane];
      XZ[0][n] = wmma_f16(a0, b, XZ[0][n]);
      XZ[1][n] = wmma_f16(a1, b, XZ[1][n]);
    }
  }
#pragma unroll
  for (int n = 0; n < 4; ++n) {
    int ncol = wc * 64 + n * 16 + lo;
    float bias = (ncol & 1) ? b_f[ncol >> 1] : b_gate[ncol >> 1];
#pragma unroll
    for (int r = 0; r < 8; ++r) { XZ[0][n][r] += bias; XZ[1][n][r] += bias; }
  }

  float taur[4];
#pragma unroll
  for (int n = 0; n < 4; ++n) taur[n] = tau[wc * 32 + n * 8 + (lo >> 1)];
  const float dt = t_eval[1] - t_eval[0];

  float hreg[32];   // [u][n][r&3]
#pragma unroll
  for (int i = 0; i < 32; ++i) hreg[i] = 0.f;

  // Opaque WH base address: redefined each iteration so LICM cannot hoist the
  // loop-invariant weight-fragment loads out of the step loop (spill hazard).
  unsigned long long whAddr = (unsigned long long)WH;

  // ---- phase 2: 64 ODE steps ----
#pragma unroll 1
  for (int t = 0; t < NSTEP; ++t) {
    asm volatile("" : "+s"(whAddr));
    const v16h* WHv = (const v16h*)whAddr;

    v8f C[2][4];
    {  // kt = 0 peeled: seed accumulators from XZ
      v16h a0 = *(const v16h*)(&lds_h[((rp * 2 + 0) * 8 + 0) * 512 + lane * 16]);
      v16h a1 = *(const v16h*)(&lds_h[((rp * 2 + 1) * 8 + 0) * 512 + lane * 16]);
#pragma unroll
      for (int n = 0; n < 4; ++n) {
        v16h b = WHv[(0 * 32 + wc * 4 + n) * 32 + lane];
        C[0][n] = wmma_f16(a0, b, XZ[0][n]);
        C[1][n] = wmma_f16(a1, b, XZ[1][n]);
      }
    }
#pragma unroll
    for (int kt = 1; kt < 8; ++kt) {
      v16h a0 = *(const v16h*)(&lds_h[((rp * 2 + 0) * 8 + kt) * 512 + lane * 16]);
      v16h a1 = *(const v16h*)(&lds_h[((rp * 2 + 1) * 8 + kt) * 512 + lane * 16]);
#pragma unroll
      for (int n = 0; n < 4; ++n) {
        v16h b = WHv[(kt * 32 + wc * 4 + n) * 32 + lane];
        C[0][n] = wmma_f16(a0, b, C[0][n]);
        C[1][n] = wmma_f16(a1, b, C[1][n]);
      }
    }
    __syncthreads();  // all A-frag reads complete before h rewrite

    // elementwise update: even lane updates r=0..3, odd lane r=4..7
#pragma unroll
    for (int n = 0; n < 4; ++n) {
      int j = wc * 32 + n * 8 + (lo >> 1);  // hidden index
      int kt2 = j >> 5, kk = j & 31;
      int h2, e2;
      if (kk < 16) { h2 = kk >> 3; e2 = kk & 7; }
      else         { int q = kk - 16; h2 = q >> 3; e2 = 8 + (q & 7); }
#pragma unroll
      for (int u = 0; u < 2; ++u) {
#pragma unroll
        for (int r = 0; r < 8; ++r) {
          float mine  = C[u][n][r];
          float other = __shfl_xor(mine, 1, 32);
          if (((r >> 2) & 1) == par) {
            float zg = par ? other : mine;
            float zf = par ? mine  : other;
            float g  = 1.f / (1.f + __expf(-zg));
            float h  = hreg[u * 16 + n * 4 + (r & 3)];
            h += dt * (zf * g - h) / taur[n];
            hreg[u * 16 + n * 4 + (r & 3)] = h;
            int row_lo = hi4 * 8 + r;                       // row within 16-block
            int lidx   = h2 * 16 + row_lo;                  // frag lane
            lds_h[((rp * 2 + u) * 8 + kt2) * 512 + lidx * 16 + e2] = (_Float16)h;
          }
        }
      }
    }
    __syncthreads();  // new h visible before next step's GEMM
  }

  // ---- phase 3: out = h @ w_out + b_out (each wave: 2 row tiles x 1 col tile) ----
  v8f Co[2] = {};
#pragma unroll
  for (int kt = 0; kt < 8; ++kt) {
    v16h a0 = *(const v16h*)(&lds_h[((rp * 2 + 0) * 8 + kt) * 512 + lane * 16]);
    v16h a1 = *(const v16h*)(&lds_h[((rp * 2 + 1) * 8 + kt) * 512 + lane * 16]);
    v16h b  = WOv[(kt * 8 + wc) * 32 + lane];
    Co[0] = wmma_f16(a0, b, Co[0]);
    Co[1] = wmma_f16(a1, b, Co[1]);
  }
  {
    int col = wc * 16 + lo;
    float bo = b_out[col];
#pragma unroll
    for (int u = 0; u < 2; ++u) {
      int rowBase = wg * MB + (rp * 2 + u) * 16;
#pragma unroll
      for (int r = 0; r < 8; ++r) {
        int row = rowBase + hi4 * 8 + r;
        out[row * OUT_ + col] = Co[u][r] + bo;
      }
    }
  }
}

extern "C" void kernel_launch(void* const* d_in, const int* in_sizes, int n_in,
                              void* d_out, int out_size, void* d_ws, size_t ws_size,
                              hipStream_t stream) {
  (void)in_sizes; (void)n_in; (void)out_size; (void)ws_size;
  const float* x      = (const float*)d_in[0];
  const float* t_eval = (const float*)d_in[1];
  const float* w_gate = (const float*)d_in[2];
  const float* b_gate = (const float*)d_in[3];
  const float* w      = (const float*)d_in[4];
  const float* b      = (const float*)d_in[5];
  const float* tau    = (const float*)d_in[6];
  const float* w_out  = (const float*)d_in[7];
  const float* b_out  = (const float*)d_in[8];
  float* out = (float*)d_out;

  char* ws = (char*)d_ws;
  _Float16* WX   = (_Float16*)(ws + WX_OFF);
  _Float16* WH   = (_Float16*)(ws + WH_OFF);
  _Float16* WOUT = (_Float16*)(ws + WOUT_OFF);
  _Float16* XF   = (_Float16*)(ws + XF_OFF);

  pack_w2 <<<256,  256, 0, stream>>>(w_gate, w, WX, 0);    // 4*32 frags
  pack_w2 <<<512,  256, 0, stream>>>(w_gate, w, WH, IN_);  // 8*32 frags
  pack_wout<<<128, 256, 0, stream>>>(w_out, WOUT);         // 8*8 frags
  pack_x  <<<1024, 256, 0, stream>>>(x, XF);               // 128*4 frags

  lnn_main<<<B_ / MB, 512, 0, stream>>>(XF, WX, WH, WOUT, t_eval, b_gate, b, tau,
                                        b_out, out);
}
